// DCNInputAdapter_21852793602798
// MI455X (gfx1250) — matrix operand order repaired and checked by reading, not compiled, not run on hardware
//
#include <hip/hip_runtime.h>
#include <hip/hip_bf16.h>

typedef __attribute__((ext_vector_type(16))) _Float16 v16h;
typedef __attribute__((ext_vector_type(8)))  float    v8f;

#define XHW 65536        // H*W
#define NBLK 8192        // (B*H*W)/128
#define NPIXF 1048576.0f // B*H*W

// ---------------------------------------------------------------------------
// Kernel 1: fused offsets + deform-sample + (DCN ⊕ residual conv) WMMA GEMM.
// One block = 128 threads = 4 waves = half an output row (b, h, 128 consecutive w).
// Writes pre-BN y into d_out and deterministic per-channel partials into ws.
// ---------------------------------------------------------------------------
__global__ __launch_bounds__(128) void dcn_fused_kernel(
    const float* __restrict__ x,      const float* __restrict__ w_off,
    const float* __restrict__ b_off,  const float* __restrict__ w_dcn,
    const float* __restrict__ w_conv, const float* __restrict__ b_conv,
    float* __restrict__ y, float* __restrict__ psum, float* __restrict__ psum2)
{
  // per-wave buffer: 32 pixels * 66 floats (stride-66 pad => conflict-free
  // column reads).  First 1024 floats double as the f16 A-row staging area.
  __shared__ float smem[4 * 2112 + 512];
  const int tid  = threadIdx.x;
  const int wv   = tid >> 5;
  const int lane = tid & 31;
  float*     wbuf  = smem + wv * 2112;
  _Float16*  aBase = (_Float16*)wbuf;          // 32 pixels * 64 halves
  float*     wOffS = smem + 4 * 2112;          // staged w_off (486) + b_off (18)

  for (int i = tid; i < 504; i += 128)
    wOffS[i] = (i < 486) ? w_off[i] : b_off[i - 486];
  __syncthreads();

  const int base = blockIdx.x * 128;           // first flat pixel of block
  const int b    = base >> 16;
  const int h    = (base >> 8) & 255;
  const int wp   = (base & 255) + tid;         // this thread's w coordinate
  const float* xb = x + b * 3 * XHW;

  // 3x3 zero-padded neighborhood (reused by offset conv AND residual im2col)
  float nb[3][9];
  #pragma unroll
  for (int t = 0; t < 9; ++t) {
    int yy = h + t / 3 - 1;
    int xx = wp + t % 3 - 1;
    bool v = (yy >= 0) && (yy < 256) && (xx >= 0) && (xx < 256);
    int idx = (yy << 8) + xx;
    nb[0][t] = v ? xb[idx]          : 0.f;
    nb[1][t] = v ? xb[XHW + idx]    : 0.f;
    nb[2][t] = v ? xb[2*XHW + idx]  : 0.f;
  }

  // offset conv: off[j] = b_off[j] + sum_{c,t} w_off[j,c,t] * nb[c][t]
  float off[18];
  for (int j = 0; j < 18; ++j) {
    float s = wOffS[486 + j];
    #pragma unroll
    for (int c = 0; c < 3; ++c)
      #pragma unroll
      for (int t = 0; t < 9; ++t)
        s = fmaf(wOffS[j*27 + c*9 + t], nb[c][t], s);
    off[j] = s;
  }

  // A-row: K 0..26 = deform taps (k*3+c), K 27..53 = im2col (27+c*9+t)
  _Float16* arow = aBase + lane * 64;
  #pragma unroll
  for (int k = 0; k < 9; ++k) {
    float py = (float)h  + (float)(k / 3 - 1) + off[2*k];
    float px = (float)wp + (float)(k % 3 - 1) + off[2*k + 1];
    float y0 = floorf(py), x0 = floorf(px);
    float ly = py - y0,   lx = px - x0;
    float hy = 1.f - ly,  hx = 1.f - lx;
    float s0 = 0.f, s1 = 0.f, s2 = 0.f;
    #pragma unroll
    for (int cr = 0; cr < 4; ++cr) {
      float yi  = y0 + (float)(cr >> 1);
      float xi  = x0 + (float)(cr & 1);
      float wgt = ((cr >> 1) ? ly : hy) * ((cr & 1) ? lx : hx);
      bool  v   = (yi >= 0.f) && (yi <= 255.f) && (xi >= 0.f) && (xi <= 255.f);
      float wv2 = v ? wgt : 0.f;
      int yc = (int)fminf(fmaxf(yi, 0.f), 255.f);
      int xc = (int)fminf(fmaxf(xi, 0.f), 255.f);
      int idx = (yc << 8) + xc;
      s0 = fmaf(wv2, xb[idx],          s0);
      s1 = fmaf(wv2, xb[XHW + idx],    s1);
      s2 = fmaf(wv2, xb[2*XHW + idx],  s2);
    }
    arow[k*3 + 0] = (_Float16)s0;
    arow[k*3 + 1] = (_Float16)s1;
    arow[k*3 + 2] = (_Float16)s2;
  }
  #pragma unroll
  for (int c = 0; c < 3; ++c)
    #pragma unroll
    for (int t = 0; t < 9; ++t)
      arow[27 + c*9 + t] = (_Float16)nb[c][t];
  #pragma unroll
  for (int k2 = 54; k2 < 64; ++k2) arow[k2] = (_Float16)0.f;
  __syncthreads();

  // B fragments of packed weights W[64K x 64N]:
  //   K<27 : W[k*3+c][o] = w_dcn[o][c][k]   (wk after reshape/transpose)
  //   27<=K<54 : W[27+c*9+t][o] = w_conv[o][c][t]
  // B layout (K=32, 16-bit): lanes 0-15 hold K=0..15 (elem e == K),
  // lanes 16-31 hold K=16..31, N = lane&15.
  v16h Bf[4][2];
  #pragma unroll
  for (int nt = 0; nt < 4; ++nt) {
    int N = nt * 16 + (lane & 15);
    #pragma unroll
    for (int ks = 0; ks < 2; ++ks) {
      #pragma unroll
      for (int e = 0; e < 16; ++e) {
        int K = ks * 32 + (lane >> 4) * 16 + e;
        float wval;
        if (K < 27)      wval = w_dcn[N*27 + (K % 3)*9 + (K / 3)];
        else if (K < 54) { int Kp = K - 27; wval = w_conv[N*27 + (Kp/9)*9 + (Kp%9)]; }
        else             wval = 0.f;
        Bf[nt][ks][e] = (_Float16)wval;
      }
    }
  }

  // A fragments in this code's contiguous K order (same as Bf):
  // lane: M=lane&15, half=lane>>4; elem e:
  // K = half*16 + e  (+32 for second K-slab)
  v16h Af[2][2];
  #pragma unroll
  for (int mt = 0; mt < 2; ++mt) {
    int M = mt * 16 + (lane & 15);
    #pragma unroll
    for (int ks = 0; ks < 2; ++ks) {
      #pragma unroll
      for (int e = 0; e < 16; ++e) {
        int j = e >> 1, pbit = e & 1;
        int K = ks*32 + (lane >> 4)*16 + e;
        Af[mt][ks][e] = aBase[M * 64 + K];
      }
    }
  }

  float biasv[4];
  #pragma unroll
  for (int nt = 0; nt < 4; ++nt) biasv[nt] = b_conv[nt * 16 + (lane & 15)];

  // 16 WMMAs per wave: 32 pixels x 64 channels, K=64 (54 real)
  #pragma unroll
  for (int mt = 0; mt < 2; ++mt) {
    #pragma unroll
    for (int nt = 0; nt < 4; ++nt) {
      v8f c;
      #pragma unroll
      for (int r = 0; r < 8; ++r) c[r] = biasv[nt];
      c = __builtin_amdgcn_wmma_f32_16x16x32_f16(false, Af[mt][0], false, Bf[nt][0],
                                                 (short)0, c, false, false);
      c = __builtin_amdgcn_wmma_f32_16x16x32_f16(false, Af[mt][1], false, Bf[nt][1],
                                                 (short)0, c, false, false);
      #pragma unroll
      for (int r = 0; r < 8; ++r) {
        int M = mt * 16 + r + ((lane >> 4) << 3);   // pixel within wave
        int N = nt * 16 + (lane & 15);              // output channel
        wbuf[M * 66 + N] = c[r];
      }
    }
  }
  __syncthreads();

  // deterministic per-block partial sums (no float atomics)
  if (tid < 64) {
    float s = 0.f, s2 = 0.f;
    for (int p = 0; p < 128; ++p) {
      float v = smem[(p >> 5) * 2112 + (p & 31) * 66 + tid];
      s += v;
      s2 = fmaf(v, v, s2);
    }
    psum [blockIdx.x * 64 + tid] = s;
    psum2[blockIdx.x * 64 + tid] = s2;
  }

  // coalesced NCHW store of pre-BN y (32 consecutive floats per wave op)
  const int wcol = (base & 255) + wv * 32 + lane;
  #pragma unroll 4
  for (int o = 0; o < 64; ++o)
    y[((b * 64 + o) << 16) + (h << 8) + wcol] = wbuf[lane * 66 + o];
}

// ---------------------------------------------------------------------------
// Kernel 2: reduce 8192 partials per channel -> BN scale/shift (deterministic)
// ---------------------------------------------------------------------------
__global__ __launch_bounds__(256) void dcn_stats_kernel(
    const float* __restrict__ psum, const float* __restrict__ psum2,
    const float* __restrict__ gamma, const float* __restrict__ beta,
    float* __restrict__ params)
{
  const int o = blockIdx.x;
  const int t = threadIdx.x;
  __shared__ float sh[256], sh2[256];
  float s = 0.f, s2 = 0.f;
  for (int i = t; i < NBLK; i += 256) {
    s  += psum [i * 64 + o];
    s2 += psum2[i * 64 + o];
  }
  sh[t] = s; sh2[t] = s2;
  __syncthreads();
  for (int k = 128; k > 0; k >>= 1) {
    if (t < k) { sh[t] += sh[t + k]; sh2[t] += sh2[t + k]; }
    __syncthreads();
  }
  if (t == 0) {
    float mean = sh[0] / NPIXF;
    float var  = sh2[0] / NPIXF - mean * mean;
    float sc   = gamma[o] * rsqrtf(var + 1e-5f);
    params[o]      = sc;
    params[64 + o] = beta[o] - mean * sc;
  }
}

// ---------------------------------------------------------------------------
// Kernel 3: in-place BN + SiLU streaming pass (float4)
// ---------------------------------------------------------------------------
__global__ __launch_bounds__(256) void dcn_bnsilu_kernel(
    float* __restrict__ y, const float* __restrict__ params)
{
  __shared__ float sc[64], sf[64];
  if (threadIdx.x < 64) {
    sc[threadIdx.x] = params[threadIdx.x];
    sf[threadIdx.x] = params[64 + threadIdx.x];
  }
  __syncthreads();
  const int total4 = 64 * XHW * 16 / 4;   // 16,777,216 float4
  for (int i = blockIdx.x * blockDim.x + threadIdx.x; i < total4;
       i += gridDim.x * blockDim.x) {
    float4 v = ((float4*)y)[i];
    int o = (i >> 14) & 63;               // 65536 floats / 4 per channel slab
    float a = sc[o], c = sf[o];
    float z;
    z = fmaf(v.x, a, c); v.x = z / (1.f + expf(-z));
    z = fmaf(v.y, a, c); v.y = z / (1.f + expf(-z));
    z = fmaf(v.z, a, c); v.z = z / (1.f + expf(-z));
    z = fmaf(v.w, a, c); v.w = z / (1.f + expf(-z));
    ((float4*)y)[i] = v;
  }
}

extern "C" void kernel_launch(void* const* d_in, const int* in_sizes, int n_in,
                              void* d_out, int out_size, void* d_ws, size_t ws_size,
                              hipStream_t stream) {
  const float* x      = (const float*)d_in[0];
  const float* w_off  = (const float*)d_in[1];
  const float* b_off  = (const float*)d_in[2];
  const float* w_dcn  = (const float*)d_in[3];
  const float* w_conv = (const float*)d_in[4];
  const float* b_conv = (const float*)d_in[5];
  const float* gamma  = (const float*)d_in[6];
  const float* beta   = (const float*)d_in[7];

  float* y      = (float*)d_out;
  float* ws     = (float*)d_ws;
  float* psum   = ws;                    // NBLK*64 floats
  float* psum2  = ws + NBLK * 64;        // NBLK*64 floats
  float* params = ws + 2 * NBLK * 64;    // 128 floats

  dcn_fused_kernel<<<NBLK, 128, 0, stream>>>(x, w_off, b_off, w_dcn, w_conv,
                                             b_conv, y, psum, psum2);
  dcn_stats_kernel<<<64, 256, 0, stream>>>(psum, psum2, gamma, beta, params);
  dcn_bnsilu_kernel<<<16384, 256, 0, stream>>>(y, params);
}
